// SupConLoss_29386166239626
// MI455X (gfx1250) — matrix operand order repaired and checked by reading, not compiled, not run on hardware
//
#include <hip/hip_runtime.h>

// ---------------------------------------------------------------------------
// SupCon loss, fused for MI455X (gfx1250), wave32 + WMMA bf16 + async-LDS.
//   N=8192 samples, D=512 features, 100 labels, temperature 0.07.
// loss = -mean_i [ (possum_i - cnt_i * lse_i) / cnt_i ]
//   possum_i = sum_{j: lbl_j==lbl_i, j!=i} sim_ij
//   lse_i    = log sum_j exp(sim_ij)   (diagonal included, row max == 1/tau)
// ---------------------------------------------------------------------------

#define NROW 8192
#define DDIM 512
#define BM   64          // rows per block
#define BN   32          // cols per tile iteration
#define NT   (NROW / BN) // 256 tile iterations
#define LDSB_STRIDE 520  // bf16 elements per LDS B row (+8 pad -> bank spread)

typedef float  v8f   __attribute__((ext_vector_type(8)));
typedef __bf16 v16bf __attribute__((ext_vector_type(16)));

union Frag {           // 16 bf16 = 32 bytes = two uint4
    uint4 q[2];
    v16bf v;
};

__device__ __forceinline__ unsigned short f2bf_rne(float f) {
    unsigned int u = __float_as_uint(f);
    unsigned int r = (u + 0x7FFFu + ((u >> 16) & 1u)) >> 16;
    return (unsigned short)r;
}

// ---- CDNA5 async global->LDS copy (ASYNCcnt path), inline asm is portable
// across ROCm 7.2 / amdgpu-toolchain builtin-arity differences.
__device__ __forceinline__ void async_b128_to_lds(unsigned lds_off, const void* g) {
    asm volatile("global_load_async_to_lds_b128 %0, %1, off"
                 :: "v"(lds_off), "v"(g) : "memory");
}
__device__ __forceinline__ void wait_async_le8() {
    asm volatile("s_wait_asynccnt 0x8" ::: "memory");
}
__device__ __forceinline__ void wait_async_le0() {
    asm volatile("s_wait_asynccnt 0x0" ::: "memory");
}

// ---------------------------------------------------------------- kernel 0
__global__ void supcon_zero_hist(int* hist) {
    if (threadIdx.x < 128) hist[threadIdx.x] = 0;
}

// ---------------------------------------------------------------- kernel 1
// One block per row: L2-normalize, convert to bf16, label histogram + u8 copy.
__global__ __launch_bounds__(256) void supcon_normalize(
    const float* __restrict__ feats, const int* __restrict__ labels,
    unsigned short* __restrict__ featsb, int* __restrict__ hist,
    unsigned char* __restrict__ lbl_u8)
{
    __shared__ float red[8];
    const int r = blockIdx.x;
    const int t = threadIdx.x;
    const float* row = feats + (size_t)r * DDIM;

    float x0 = row[t];
    float x1 = row[t + 256];
    float ss = x0 * x0 + x1 * x1;
#pragma unroll
    for (int m = 1; m < 32; m <<= 1) ss += __shfl_xor(ss, m, 32);
    if ((t & 31) == 0) red[t >> 5] = ss;
    __syncthreads();
    if (t == 0) {
        float tot = 0.f;
#pragma unroll
        for (int w = 0; w < 8; ++w) tot += red[w];
        red[0] = tot;
    }
    __syncthreads();
    const float inv = rsqrtf(red[0]);

    featsb[(size_t)r * DDIM + t]       = f2bf_rne(x0 * inv);
    featsb[(size_t)r * DDIM + t + 256] = f2bf_rne(x1 * inv);

    if (t == 0) {
        const int l = labels[r];
        atomicAdd(&hist[l], 1);
        lbl_u8[r] = (unsigned char)l;
    }
}

// ---------------------------------------------------------------- kernel 2
// Fused sim-GEMM (bf16 WMMA) + streaming exp-sum + positive-pair sum.
// Block: 256 threads = 8 waves, wave grid 4(wm) x 2(wn); block owns 64 rows.
// B tiles double-buffered in LDS via async global->LDS loads.
__global__ __launch_bounds__(256) void supcon_main(
    const unsigned short* __restrict__ featsb, const int* __restrict__ labels,
    const unsigned char* __restrict__ lbl_u8, const int* __restrict__ hist,
    float* __restrict__ per_row)
{
    __shared__ __align__(16) unsigned short Bs[2 * BN * LDSB_STRIDE]; // 66,560 B
    __shared__ __align__(16) unsigned char  Ls[NROW];                 //  8,192 B
    __shared__ float sred[BM][2];
    __shared__ float pred[BM][2];

    const int tid  = threadIdx.x;
    const int lane = tid & 31;
    const int wave = tid >> 5;     // 0..7
    const int wm   = wave >> 1;    // 0..3 : 16-row stripe
    const int wn   = wave & 1;     // 0..1 : 16-col stripe within tile
    const int half = lane >> 4;    // 0/1 half-wave
    const int l16  = lane & 15;

    const int block_row = blockIdx.x * BM;
    const int row0      = block_row + wm * 16;

    const float INVT = 1.0f / 0.07f;
    const float MCAP = INVT;       // exact row max (diagonal of normalized sim)

    const unsigned ls_base = (unsigned)(size_t)&Ls[0];
    const unsigned bs_base = (unsigned)(size_t)&Bs[0];

    // ---- async-stage all 8192 label bytes into LDS (2 x b128 per thread)
#pragma unroll
    for (int rep = 0; rep < 2; ++rep) {
        const int idx = tid + 256 * rep;              // 512 x 16B = 8 KiB
        async_b128_to_lds(ls_base + idx * 16, (const void*)(lbl_u8 + idx * 16));
    }

    // ---- async-issue B tile 0 into buffer 0 (8 x b128 per thread)
    {
        const unsigned char* g0 = (const unsigned char*)featsb;  // tile 0 base
#pragma unroll
        for (int rep = 0; rep < 8; ++rep) {
            const int idx = tid + 256 * rep;          // 2048 x 16B per tile
            const int rr  = idx >> 6;                 // feature row in tile
            const int cc  = idx & 63;                 // 16B chunk in row
            async_b128_to_lds(bs_base + (rr * 65 + cc) * 16,
                              (const void*)(g0 + (size_t)rr * 1024 + cc * 16));
        }
    }

    // ---- preload this wave's A stripe: 16 rows x K=512 bf16 in registers.
    // A-frag layout (16x32 bf16): lane = M%16; half 0 holds K {k0..k0+7},
    // {k0+16..k0+23}; half 1 the +8 shifted chunks (ISA 7.12.2).
    Frag a[16];
    const unsigned short* aptr = featsb + (size_t)(row0 + l16) * DDIM;
#pragma unroll
    for (int kk = 0; kk < 16; ++kk) {
        const int k0 = kk * 32 + half * 8;
        a[kk].q[0] = *(const uint4*)(aptr + k0);
        a[kk].q[1] = *(const uint4*)(aptr + k0 + 16);
    }

    // labels of the 8 rows this lane's C-fragment slots map to (one-time)
    int lblr[8];
#pragma unroll
    for (int i = 0; i < 8; ++i) lblr[i] = labels[row0 + half * 8 + i];

    float s[8], p[8];
#pragma unroll
    for (int i = 0; i < 8; ++i) { s[i] = 0.f; p[i] = 0.f; }

    for (int it = 0; it < NT; ++it) {
        const int cur     = it & 1;
        const int colbase = it * BN;

        // ---- async-prefetch next tile into the other buffer
        if (it + 1 < NT) {
            const unsigned char* g =
                (const unsigned char*)(featsb + (size_t)(colbase + BN) * DDIM);
            const unsigned dst = bs_base + (unsigned)(cur ^ 1) * (BN * LDSB_STRIDE * 2);
#pragma unroll
            for (int rep = 0; rep < 8; ++rep) {
                const int idx = tid + 256 * rep;
                const int rr  = idx >> 6;
                const int cc  = idx & 63;
                async_b128_to_lds(dst + (rr * 65 + cc) * 16,
                                  (const void*)(g + (size_t)rr * 1024 + cc * 16));
            }
            wait_async_le8();   // in-order: tile `it` (and labels) now in LDS
        } else {
            wait_async_le0();
        }
        __syncthreads();        // all waves' copies of tile `it` visible

        // ---- 16x16x512 accumulation via 16 chained WMMAs
        v8f acc = {0.f, 0.f, 0.f, 0.f, 0.f, 0.f, 0.f, 0.f};
        // B-frag layout (32x16 bf16): lane = N%16; half 0 holds K {0..7},{16..23};
        // half 1 the +8 shifted chunks (same as A).
        const unsigned short* bbase = Bs + (size_t)cur * (BN * LDSB_STRIDE)
                                    + (size_t)(wn * 16 + l16) * LDSB_STRIDE
                                    + half * 8;
#pragma unroll
        for (int kk = 0; kk < 16; ++kk) {
            Frag b;
            b.q[0] = *(const uint4*)(bbase + kk * 32);
            b.q[1] = *(const uint4*)(bbase + kk * 32 + 16);
            acc = __builtin_amdgcn_wmma_f32_16x16x32_bf16(
                false, a[kk].v, false, b.v, (short)0, acc, false, false);
        }

        // ---- fused epilogue for this 16x16 tile (labels from LDS, no vmem)
        const int C    = colbase + wn * 16 + l16;
        const int lblc = (int)Ls[C];
#pragma unroll
        for (int i = 0; i < 8; ++i) {
            const float v = acc[i] * INVT;              // sim value
            float e  = __expf(v - MCAP);                // capped exp (<= ~1)
            const int R = row0 + half * 8 + i;
            float pv = (lblc == lblr[i] && C != R) ? v : 0.f;
#pragma unroll
            for (int m = 1; m < 16; m <<= 1) {          // reduce 16 columns
                e  += __shfl_xor(e, m, 32);
                pv += __shfl_xor(pv, m, 32);
            }
            s[i] += e;
            p[i] += pv;
        }
        __syncthreads();        // reads of buf `cur` done before next overwrite
    }

    // ---- combine the two column-waves per row, emit mean_log_prob_pos
    if (l16 == 0) {
#pragma unroll
        for (int i = 0; i < 8; ++i) {
            const int rl = wm * 16 + half * 8 + i;
            sred[rl][wn] = s[i];
            pred[rl][wn] = p[i];
        }
    }
    __syncthreads();
    if (tid < BM) {
        const float st  = sred[tid][0] + sred[tid][1];
        const float pt  = pred[tid][0] + pred[tid][1];
        const int   R   = block_row + tid;
        const float cnt = (float)(hist[labels[R]] - 1);
        const float lse = MCAP + __logf(st);
        per_row[R] = (pt - cnt * lse) / cnt;
    }
}

// ---------------------------------------------------------------- kernel 3
__global__ __launch_bounds__(256) void supcon_reduce(
    const float* __restrict__ per_row, float* __restrict__ out)
{
    __shared__ float red[256];
    const int t = threadIdx.x;
    float sum = 0.f;
#pragma unroll
    for (int k = 0; k < NROW / 256; ++k) sum += per_row[t + 256 * k];
    red[t] = sum;
    __syncthreads();
#pragma unroll
    for (int sft = 128; sft > 0; sft >>= 1) {
        if (t < sft) red[t] += red[t + sft];
        __syncthreads();
    }
    if (t == 0) out[0] = -red[0] / (float)NROW;
}

// ---------------------------------------------------------------------------
extern "C" void kernel_launch(void* const* d_in, const int* in_sizes, int n_in,
                              void* d_out, int out_size, void* d_ws, size_t ws_size,
                              hipStream_t stream) {
    const float* feats  = (const float*)d_in[0];
    const int*   labels = (const int*)d_in[1];

    // workspace layout
    unsigned short* featsb  = (unsigned short*)d_ws;                       // 8 MiB
    int*            hist    = (int*)((char*)d_ws + (size_t)NROW * DDIM * 2);
    unsigned char*  lbl_u8  = (unsigned char*)((char*)hist + 128 * sizeof(int));
    float*          per_row = (float*)((char*)lbl_u8 + NROW);

    supcon_zero_hist<<<1, 128, 0, stream>>>(hist);
    supcon_normalize<<<NROW, 256, 0, stream>>>(feats, labels, featsb, hist, lbl_u8);
    supcon_main<<<NROW / BM, 256, 0, stream>>>(featsb, labels, lbl_u8, hist, per_row);
    supcon_reduce<<<1, 256, 0, stream>>>(per_row, (float*)d_out);
}
